// VGCN_Link_28346784154173
// MI455X (gfx1250) — matrix operand order repaired and checked by reading, not compiled, not run on hardware
//
#include <hip/hip_runtime.h>
#include <math.h>

typedef __attribute__((ext_vector_type(16))) _Float16 v16h;
typedef __attribute__((ext_vector_type(8)))  _Float16 v8h;
typedef __attribute__((ext_vector_type(8)))  float    v8f;

constexpr int N_NODES = 10000;
constexpr int E_EDGES = 320000;
constexpr int F_IN    = 512;
constexpr int H_DIM   = 256;
constexpr int Z_DIM   = 64;

// ---------------- elementwise helpers ----------------

__global__ void k_cast_f16(const float* __restrict__ in, _Float16* __restrict__ out, int n) {
    int i = blockIdx.x * blockDim.x + threadIdx.x;
    if (i < n) out[i] = (_Float16)in[i];
}

// out[c*R + r] = (f16) in[r*C + c]   (transpose-cast, launched over R*C)
__global__ void k_transpose_f16(const float* __restrict__ in, _Float16* __restrict__ out,
                                int R, int C) {
    int i = blockIdx.x * blockDim.x + threadIdx.x;
    if (i >= R * C) return;
    int c = i % C, r = i / C;
    out[(size_t)c * R + r] = (_Float16)in[i];
}

// pack Wm[256,64], Ws[256,64] into Wct[128][256]: row n holds column n of Wm (n<64) or Ws
__global__ void k_pack_wc(const float* __restrict__ Wm, const float* __restrict__ Ws,
                          _Float16* __restrict__ out) {
    int i = blockIdx.x * blockDim.x + threadIdx.x;   // over 128*256
    if (i >= 128 * 256) return;
    int k = i % 256, n = i / 256;
    float v = (n < Z_DIM) ? Wm[(size_t)k * Z_DIM + n] : Ws[(size_t)k * Z_DIM + (n - Z_DIM)];
    out[(size_t)n * 256 + k] = (_Float16)v;
}

__global__ void k_bias_relu_f16(const float* __restrict__ agg, const float* __restrict__ b,
                                _Float16* __restrict__ out, int n, int F) {
    int i = blockIdx.x * blockDim.x + threadIdx.x;
    if (i >= n) return;
    float v = agg[i] + b[i % F];
    out[i] = (_Float16)(v > 0.0f ? v : 0.0f);
}

// z = eps * exp(logstd) + mean ; agg2 rows hold [mean(64) | logstd(64)]
__global__ void k_reparam(const float* __restrict__ agg2, const float* __restrict__ bm,
                          const float* __restrict__ bs, const float* __restrict__ eps,
                          _Float16* __restrict__ zh) {
    int i = blockIdx.x * blockDim.x + threadIdx.x;
    if (i >= N_NODES * Z_DIM) return;
    int n = i / Z_DIM, c = i % Z_DIM;
    float mean = agg2[(size_t)n * 128 + c]      + bm[c];
    float ls   = agg2[(size_t)n * 128 + 64 + c] + bs[c];
    zh[i] = (_Float16)(eps[i] * __expf(ls) + mean);
}

// ---------------- SpMM edge scatter: agg[dst] += support[src] * w ----------------
// One thread handles 4 consecutive features of one edge. F4 = F/4.
__global__ void k_edge_scatter(const float* __restrict__ sup, const int* __restrict__ src,
                               const int* __restrict__ dst, const float* __restrict__ w,
                               float* __restrict__ agg, int F4) {
    int i = blockIdx.x * blockDim.x + threadIdx.x;
    int e = i / F4;
    if (e >= E_EDGES) return;
    int f = (i % F4) * 4;
    int s = src[e], d = dst[e];
    float wt = w[e];
    const int F = F4 * 4;
    const float4 v = *(const float4*)(sup + (size_t)s * F + f);
    float* p = agg + (size_t)d * F + f;
    unsafeAtomicAdd(p + 0, v.x * wt);
    unsafeAtomicAdd(p + 1, v.y * wt);
    unsafeAtomicAdd(p + 2, v.z * wt);
    unsafeAtomicAdd(p + 3, v.w * wt);
}

// ---------------- WMMA GEMM: C[M,Nc] = A[M,K] * Bt[Nc,K]^T ----------------
// f16 inputs, f32 accumulate, one wave per 16x16 C tile.
// Per-lane layouts used here (contiguous k, same order for A and B; not the ISA table order):
//   A (16x32 f16): lane l, row M = l&15; halves i = K kb+i,
//                  kb = (l<16) ? 0 : 16
//   B (32x16 f16): lane l, col N = l&15; halves i = K kb2+i, kb2 = (l<16) ? 0 : 16
//   C/D (f32):     VGPR r: lanes 0..15 -> M=r, lanes 16..31 -> M=r+8; N = lane&15
template <bool SIGMOID>
__global__ __launch_bounds__(256)
void k_wmma_gemm(const _Float16* __restrict__ A, const _Float16* __restrict__ Bt,
                 float* __restrict__ C, int M, int Nc, int K) {
    const int lane = threadIdx.x & 31;
    const int wid  = (blockIdx.x * blockDim.x + threadIdx.x) >> 5;   // wave-uniform
    const int tiles_n = Nc >> 4;
    const int total   = (M >> 4) * tiles_n;
    if (wid >= total) return;                                        // uniform branch: EXEC stays full
    const int ti = wid / tiles_n;
    const int tj = wid % tiles_n;

    const int lh   = lane & 15;
    const int asel = (lane < 16) ? 0 : 16;
    const int bsel = (lane < 16) ? 0 : 16;
    const _Float16* arow = A  + (size_t)(ti * 16 + lh) * K;
    const _Float16* brow = Bt + (size_t)(tj * 16 + lh) * K;

    v8f acc = {};
    for (int kk = 0; kk < K; kk += 32) {
        v8h alo = *(const v8h*)(arow + kk + asel);
        v8h ahi = *(const v8h*)(arow + kk + asel + 8);
        v16h a, b;
        b = *(const v16h*)(brow + kk + bsel);
#pragma unroll
        for (int q = 0; q < 8; ++q) { a[q] = alo[q]; a[8 + q] = ahi[q]; }
        acc = __builtin_amdgcn_wmma_f32_16x16x32_f16(
            /*neg_a=*/false, a, /*neg_b=*/false, b,
            /*c_mod=*/(short)0, acc, /*reuse_a=*/false, /*reuse_b=*/false);
    }

    const int col   = tj * 16 + lh;
    const int rbase = ti * 16 + ((lane < 16) ? 0 : 8);
#pragma unroll
    for (int r = 0; r < 8; ++r) {
        float v = acc[r];
        if (SIGMOID) v = 1.0f / (1.0f + __expf(-v));
        C[(size_t)(rbase + r) * Nc + col] = v;
    }
}

// ---------------- host-side orchestration ----------------

extern "C" void kernel_launch(void* const* d_in, const int* in_sizes, int n_in,
                              void* d_out, int out_size, void* d_ws, size_t ws_size,
                              hipStream_t stream) {
    const float* x    = (const float*)d_in[0];
    const int*   esrc = (const int*)  d_in[1];
    const int*   edst = (const int*)  d_in[2];
    const float* ew   = (const float*)d_in[3];
    const float* eps  = (const float*)d_in[4];
    const float* W1   = (const float*)d_in[5];
    const float* b1   = (const float*)d_in[6];
    const float* Wm   = (const float*)d_in[7];
    const float* bm   = (const float*)d_in[8];
    const float* Ws   = (const float*)d_in[9];
    const float* bs   = (const float*)d_in[10];
    float* out = (float*)d_out;

    // workspace layout (all offsets multiples of 256 B)
    char* ws = (char*)d_ws;
    size_t off = 0;
    _Float16* xh   = (_Float16*)(ws + off); off += (size_t)N_NODES * F_IN * 2;   // 10.24 MB
    _Float16* w1t  = (_Float16*)(ws + off); off += (size_t)F_IN * H_DIM * 2;     // 0.26 MB  [256,512]
    _Float16* wct  = (_Float16*)(ws + off); off += (size_t)128 * H_DIM * 2;      // 64 KB    [128,256]
    float*    sup1 = (float*)   (ws + off); off += (size_t)N_NODES * H_DIM * 4;  // 10.24 MB
    float*    agg1 = (float*)   (ws + off); off += (size_t)N_NODES * H_DIM * 4;  // 10.24 MB
    _Float16* hh   = (_Float16*)(ws + off); off += (size_t)N_NODES * H_DIM * 2;  // 5.12 MB
    float*    sup2 = (float*)   (ws + off); off += (size_t)N_NODES * 128 * 4;    // 5.12 MB
    float*    agg2 = (float*)   (ws + off); off += (size_t)N_NODES * 128 * 4;    // 5.12 MB
    _Float16* zh   = (_Float16*)(ws + off); off += (size_t)N_NODES * Z_DIM * 2;  // 1.28 MB
    (void)ws_size; (void)n_in; (void)in_sizes; (void)out_size;

    const int B = 256;

    // zero scatter-add accumulators (memset nodes are graph-capturable)
    hipMemsetAsync(agg1, 0, (size_t)N_NODES * H_DIM * 4, stream);
    hipMemsetAsync(agg2, 0, (size_t)N_NODES * 128 * 4, stream);

    // casts / repacks
    {
        int n = N_NODES * F_IN;
        k_cast_f16<<<(n + B - 1) / B, B, 0, stream>>>(x, xh, n);
    }
    k_transpose_f16<<<(F_IN * H_DIM + B - 1) / B, B, 0, stream>>>(W1, w1t, F_IN, H_DIM);
    k_pack_wc<<<(128 * 256 + B - 1) / B, B, 0, stream>>>(Wm, Ws, wct);

    // layer 1: sup1 = xh @ W1  (C tiles: 625 x 16 = 10000 waves)
    k_wmma_gemm<false><<<(10000 + 7) / 8, B, 0, stream>>>(xh, w1t, sup1, N_NODES, H_DIM, F_IN);

    // SpMM 1: agg1[dst] += sup1[src] * w   (E * 64 quad-feature threads)
    k_edge_scatter<<<(E_EDGES * (H_DIM / 4) + B - 1) / B, B, 0, stream>>>(
        sup1, esrc, edst, ew, agg1, H_DIM / 4);

    // hidden = relu(agg1 + b1) -> f16
    {
        int n = N_NODES * H_DIM;
        k_bias_relu_f16<<<(n + B - 1) / B, B, 0, stream>>>(agg1, b1, hh, n, H_DIM);
    }

    // layer 2 (fused Wm|Ws): sup2 = hh @ Wc  (625 x 8 = 5000 waves)
    k_wmma_gemm<false><<<(5000 + 7) / 8, B, 0, stream>>>(hh, wct, sup2, N_NODES, 128, H_DIM);

    // SpMM 2: agg2[dst] += sup2[src] * w
    k_edge_scatter<<<(E_EDGES * (128 / 4) + B - 1) / B, B, 0, stream>>>(
        sup2, esrc, edst, ew, agg2, 128 / 4);

    // z = eps * exp(logstd + bs) ... actually logstd = agg2[:,64:128]+bs ; mean = agg2[:,0:64]+bm
    {
        int n = N_NODES * Z_DIM;
        k_reparam<<<(n + B - 1) / B, B, 0, stream>>>(agg2, bm, bs, eps, zh);
    }

    // decode: out = sigmoid(z @ z^T)  (625 x 625 = 390625 tiles)
    {
        long long tiles = (long long)(N_NODES / 16) * (N_NODES / 16);
        int blocks = (int)((tiles + 7) / 8);
        k_wmma_gemm<true><<<blocks, B, 0, stream>>>(zh, zh, out, N_NODES, N_NODES, Z_DIM);
    }
}
